// MambaModel_31456340475917
// MI455X (gfx1250) — matrix operand order repaired and checked by reading, not compiled, not run on hardware
//
#include <hip/hip_runtime.h>
#include <hip/hip_bf16.h>
#include <math.h>

// ---------------- model dims ----------------
#define VOCAB      128
#define D_MODEL    1024
#define D_STATE    128
#define D_CONV     4
#define N_LAYERS   4
#define HEADDIM    64
#define D_INNER    2048
#define NHEADS     32
#define CONV_DIM   (D_INNER + 2 * D_STATE)       // 2304
#define D_IN_PROJ  (2 * D_INNER + 2 * D_STATE + NHEADS) // 4384
#define ZXW        4416                          // D_IN_PROJ padded to 64-col multiple
#define BATCH      2
#define SEQ        2048
#define MROWS      (BATCH * SEQ)                 // 4096

typedef __attribute__((ext_vector_type(16))) __bf16 v16bf;
typedef __attribute__((ext_vector_type(8)))  __bf16 v8bf;
typedef __attribute__((ext_vector_type(8)))  float  v8f;

// ---------------- helpers ----------------
__device__ __forceinline__ unsigned short f2bf(float f) {
    unsigned u = __float_as_uint(f);
    u += 0x7fffu + ((u >> 16) & 1u);             // round-to-nearest-even
    return (unsigned short)(u >> 16);
}
__device__ __forceinline__ float silu_f(float x) { return x / (1.f + __expf(-x)); }

// A fragment: 16x32 bf16, row = lane&15, K halves split per ISA layout:
// lanes 0-15: K = {k0..k0+7, k0+16..k0+23}; lanes 16-31: +8.
__device__ __forceinline__ v16bf load_frag_a(const unsigned short* base, int ld,
                                             int row0, int k0) {
    int lane = threadIdx.x & 31;
    int r = lane & 15, hf = lane >> 4;
    const __bf16* p = (const __bf16*)(base + (size_t)(row0 + r) * ld + k0 + hf * 8);
    v8bf lo = *reinterpret_cast<const v8bf*>(p);
    v8bf hi = *reinterpret_cast<const v8bf*>(p + 16);
    v16bf f;
#pragma unroll
    for (int i = 0; i < 8; ++i) { f[i] = lo[i]; f[i + 8] = hi[i]; }
    return f;
}
// B fragment: 32x16 (K x N) from row-major W[N,K]: col n = lane&15,
// lanes 0-15: K = {k0..k0+7, k0+16..k0+23}; lanes 16-31: +8.
__device__ __forceinline__ v16bf load_frag_b(const unsigned short* base, int ld,
                                             int n0, int k0) {
    int lane = threadIdx.x & 31;
    int n = lane & 15, hf = lane >> 4;
    const __bf16* p = (const __bf16*)(base + (size_t)(n0 + n) * ld + k0 + hf * 8);
    v8bf lo = *reinterpret_cast<const v8bf*>(p);
    v8bf hi = *reinterpret_cast<const v8bf*>(p + 16);
    v16bf f;
#pragma unroll
    for (int i = 0; i < 8; ++i) { f[i] = lo[i]; f[i + 8] = hi[i]; }
    return f;
}

#define WMMA_BF16(a, b, c) \
    __builtin_amdgcn_wmma_f32_16x16x32_bf16(false, (a), false, (b), (short)0, (c), false, false)

// ---------------- kernels ----------------
__global__ void embed_kernel(const int* __restrict__ tok,
                             const float* __restrict__ emb,
                             float* __restrict__ x) {
    int row = blockIdx.x;
    int t = tok[row];
    const float* src = emb + (size_t)t * D_MODEL;
    float* dst = x + (size_t)row * D_MODEL;
    for (int i = threadIdx.x; i < D_MODEL; i += 256) dst[i] = src[i];
}

// convert n fp32 -> bf16, zero-fill up to npad elements
__global__ void cvt_bf16_pad_kernel(const float* __restrict__ src,
                                    unsigned short* __restrict__ dst, int n, int npad) {
    int i = blockIdx.x * 256 + threadIdx.x;
    if (i < npad) dst[i] = (i < n) ? f2bf(src[i]) : (unsigned short)0;
}

// rmsnorm(x,w) -> bf16, one block per row
__global__ void rmsnorm_bf16_kernel(const float* __restrict__ X,
                                    const float* __restrict__ w,
                                    unsigned short* __restrict__ out, int D) {
    int row = blockIdx.x, t = threadIdx.x;
    const float* xr = X + (size_t)row * D;
    float ss = 0.f;
    for (int i = t; i < D; i += 256) { float v = xr[i]; ss += v * v; }
    __shared__ float red[256];
    red[t] = ss; __syncthreads();
    for (int s = 128; s > 0; s >>= 1) { if (t < s) red[t] += red[t + s]; __syncthreads(); }
    float scale = rsqrtf(red[0] / (float)D + 1e-5f);
    unsigned short* o = out + (size_t)row * D;
    for (int i = t; i < D; i += 256) o[i] = f2bf(xr[i] * scale * w[i]);
}

// rmsnorm(y*silu(z), w) -> bf16, D = D_INNER, one block/row, 8 vals/thread
__global__ void gated_rmsnorm_bf16_kernel(const float* __restrict__ Y,
                                          const float* __restrict__ ZX, // rows of width ZXW, z = first D_INNER cols
                                          const float* __restrict__ w,
                                          unsigned short* __restrict__ out) {
    int row = blockIdx.x, t = threadIdx.x;
    const float* yr = Y + (size_t)row * D_INNER;
    const float* zr = ZX + (size_t)row * ZXW;
    float v[8]; float ss = 0.f;
#pragma unroll
    for (int j = 0; j < 8; ++j) {
        int i = t + j * 256;
        float g = yr[i] * silu_f(zr[i]);
        v[j] = g; ss += g * g;
    }
    __shared__ float red[256];
    red[t] = ss; __syncthreads();
    for (int s = 128; s > 0; s >>= 1) { if (t < s) red[t] += red[t + s]; __syncthreads(); }
    float scale = rsqrtf(red[0] / (float)D_INNER + 1e-5f);
    unsigned short* o = out + (size_t)row * D_INNER;
#pragma unroll
    for (int j = 0; j < 8; ++j) { int i = t + j * 256; o[i] = f2bf(v[j] * scale * w[i]); }
}

// C[M,N] = A[M,K](bf16) * W[N,K]^T(bf16) (+ Res), WMMA bf16 16x16x32.
// block: 256 thr = 8 waves in a 4(M) x 2(N) grid; block tile 128 x 64.
// wave tile: 32x32 = 2x2 WMMA tiles -> 4 WMMAs per 4 fragment loads per K-step.
__global__ void gemm_bf16_wmma_kernel(const unsigned short* __restrict__ A,
                                      const unsigned short* __restrict__ W,
                                      float* __restrict__ C,
                                      const float* __restrict__ Res,
                                      int M, int N, int K) {
    int wid = threadIdx.x >> 5;
    int wm = wid & 3, wn = wid >> 2;
    int mbase = blockIdx.y * 128 + wm * 32;
    int nbase = blockIdx.x * 64 + wn * 32;
    v8f acc00 = {}, acc01 = {}, acc10 = {}, acc11 = {};
#pragma unroll 2
    for (int k = 0; k < K; k += 32) {
        v16bf a0 = load_frag_a(A, K, mbase,      k);
        v16bf a1 = load_frag_a(A, K, mbase + 16, k);
        v16bf b0 = load_frag_b(W, K, nbase,      k);
        v16bf b1 = load_frag_b(W, K, nbase + 16, k);
        acc00 = WMMA_BF16(a0, b0, acc00);
        acc01 = WMMA_BF16(a0, b1, acc01);
        acc10 = WMMA_BF16(a1, b0, acc10);
        acc11 = WMMA_BF16(a1, b1, acc11);
    }
    int lane = threadIdx.x & 31;
    int half = lane >> 4, col = lane & 15;
#pragma unroll
    for (int r = 0; r < 8; ++r) {
        int row0 = mbase + r + 8 * half;
        int row1 = row0 + 16;
        size_t o00 = (size_t)row0 * N + nbase + col;
        size_t o01 = o00 + 16;
        size_t o10 = (size_t)row1 * N + nbase + col;
        size_t o11 = o10 + 16;
        float v00 = acc00[r], v01 = acc01[r], v10 = acc10[r], v11 = acc11[r];
        if (Res) { v00 += Res[o00]; v01 += Res[o01]; v10 += Res[o10]; v11 += Res[o11]; }
        C[o00] = v00; C[o01] = v01; C[o10] = v10; C[o11] = v11;
    }
}

// depthwise causal conv (4 taps) + bias + silu over CONV_DIM channels
__global__ void conv_silu_kernel(const float* __restrict__ zx,
                                 const float* __restrict__ cw,
                                 const float* __restrict__ cb,
                                 float* __restrict__ xbc) {
    int c = blockIdx.x * 256 + threadIdx.x;
    int row = blockIdx.y;
    if (c >= CONV_DIM) return;
    int b = row / SEQ, l = row % SEQ;
    float acc = cb[c];
#pragma unroll
    for (int k = 0; k < D_CONV; ++k) {
        int li = l + k - (D_CONV - 1);
        if (li >= 0)
            acc += zx[((size_t)(b * SEQ + li)) * ZXW + D_INNER + c] * cw[c * D_CONV + k];
    }
    xbc[(size_t)row * CONV_DIM + c] = silu_f(acc);
}

__global__ void dt_softplus_kernel(const float* __restrict__ zx,
                                   const float* __restrict__ dtb,
                                   float* __restrict__ dt) {
    int i = blockIdx.x * 256 + threadIdx.x;
    if (i >= MROWS * NHEADS) return;
    int row = i / NHEADS, h = i - row * NHEADS;
    float v = zx[(size_t)row * ZXW + D_INNER + CONV_DIM + h] + dtb[h];
    dt[i] = (v > 20.f) ? v : log1pf(__expf(v));
}

// sequential selective scan: one block per (batch, head), 256 thr = 8 waves.
// wave owns 8 of 64 P rows; lane owns 4 of 128 N states -> 32 states/thread.
__global__ void scan_kernel(const float* __restrict__ xbc,
                            const float* __restrict__ dtm,
                            const float* __restrict__ alog,
                            const float* __restrict__ dpar,
                            float* __restrict__ y) {
    int b = blockIdx.x >> 5;
    int h = blockIdx.x & 31;
    float A  = -__expf(alog[h]);
    float Dp = dpar[h];
    int t = threadIdx.x;
    int wave = t >> 5;   // p block
    int lane = t & 31;   // n block
    float hs[8][4];
#pragma unroll
    for (int i = 0; i < 8; ++i)
#pragma unroll
        for (int j = 0; j < 4; ++j) hs[i][j] = 0.f;
    __shared__ float sx[HEADDIM], sB[D_STATE], sC[D_STATE];
    for (int l = 0; l < SEQ; ++l) {
        size_t row = (size_t)b * SEQ + l;
        const float* xr = xbc + row * CONV_DIM;
        if (t < 64)       sx[t]        = xr[h * HEADDIM + t];
        else if (t < 192) sB[t - 64]   = xr[D_INNER + (t - 64)];
        else              sC[t - 192]  = xr[D_INNER + D_STATE + (t - 192)];
        if (t < 64)       sC[64 + t]   = xr[D_INNER + D_STATE + 64 + t];
        float dtv = dtm[row * NHEADS + h];
        float dA  = __expf(dtv * A);
        __syncthreads();
        float yp[8];
#pragma unroll
        for (int i = 0; i < 8; ++i) {
            float xv = dtv * sx[wave * 8 + i];
            float acc = 0.f;
#pragma unroll
            for (int j = 0; j < 4; ++j) {
                int n = lane * 4 + j;
                hs[i][j] = hs[i][j] * dA + xv * sB[n];
                acc += hs[i][j] * sC[n];
            }
            yp[i] = acc;
        }
        // full-wave reduce over the 128 N states (wave32)
#pragma unroll
        for (int i = 0; i < 8; ++i)
            for (int off = 16; off; off >>= 1) yp[i] += __shfl_xor(yp[i], off, 32);
        if (lane == 0) {
#pragma unroll
            for (int i = 0; i < 8; ++i) {
                int p = wave * 8 + i;
                y[row * (size_t)D_INNER + h * HEADDIM + p] = yp[i] + Dp * sx[p];
            }
        }
        __syncthreads();
    }
}

// ---------------- workspace layout (bytes) ----------------
#define OFF_X    0ull
#define OFF_XN   (OFF_X   + (size_t)MROWS * D_MODEL * 4)
#define OFF_WBF  (OFF_XN  + (size_t)MROWS * D_MODEL * 2)      // bf16 weights (max ZXW*1024)
#define OFF_ZX   (OFF_WBF + (size_t)ZXW * D_MODEL * 2)
#define OFF_XBC  (OFF_ZX  + (size_t)MROWS * ZXW * 4)
#define OFF_DT   (OFF_XBC + (size_t)MROWS * CONV_DIM * 4)
#define OFF_Y    (OFF_DT  + (size_t)MROWS * NHEADS * 4)
#define OFF_YZ   (OFF_Y   + (size_t)MROWS * D_INNER * 4)

extern "C" void kernel_launch(void* const* d_in, const int* in_sizes, int n_in,
                              void* d_out, int out_size, void* d_ws, size_t ws_size,
                              hipStream_t stream) {
    const int*   tokens   = (const int*)  d_in[0];
    const float* emb      = (const float*)d_in[1];
    const float* in_w     = (const float*)d_in[2];
    const float* conv_w   = (const float*)d_in[3];
    const float* conv_b   = (const float*)d_in[4];
    const float* dt_bias  = (const float*)d_in[5];
    const float* A_log    = (const float*)d_in[6];
    const float* D_param  = (const float*)d_in[7];
    const float* mnw      = (const float*)d_in[8];
    const float* out_w    = (const float*)d_in[9];
    const float* norm_w   = (const float*)d_in[10];
    const float* norm_f_w = (const float*)d_in[11];
    const float* lm_head  = (const float*)d_in[12];

    char* ws = (char*)d_ws;
    float*          x   = (float*)(ws + OFF_X);
    unsigned short* xn  = (unsigned short*)(ws + OFF_XN);
    unsigned short* wbf = (unsigned short*)(ws + OFF_WBF);
    float*          zx  = (float*)(ws + OFF_ZX);
    float*          xbc = (float*)(ws + OFF_XBC);
    float*          dtb = (float*)(ws + OFF_DT);
    float*          yb  = (float*)(ws + OFF_Y);
    unsigned short* yz  = (unsigned short*)(ws + OFF_YZ);

    dim3 blk(256);

    embed_kernel<<<MROWS, blk, 0, stream>>>(tokens, emb, x);

    for (int l = 0; l < N_LAYERS; ++l) {
        // xn = rmsnorm(x, norm_w[l]) -> bf16
        rmsnorm_bf16_kernel<<<MROWS, blk, 0, stream>>>(x, norm_w + l * D_MODEL, xn, D_MODEL);

        // zx = xn @ in_w[l]^T   (N padded 4384 -> 4416, pad weight rows zeroed)
        int nW = D_IN_PROJ * D_MODEL, nWp = ZXW * D_MODEL;
        cvt_bf16_pad_kernel<<<(nWp + 255) / 256, blk, 0, stream>>>(
            in_w + (size_t)l * nW, wbf, nW, nWp);
        gemm_bf16_wmma_kernel<<<dim3(ZXW / 64, MROWS / 128), blk, 0, stream>>>(
            xn, wbf, zx, nullptr, MROWS, ZXW, D_MODEL);

        // conv + silu, softplus(dt)
        conv_silu_kernel<<<dim3((CONV_DIM + 255) / 256, MROWS), blk, 0, stream>>>(
            zx, conv_w + (size_t)l * CONV_DIM * D_CONV, conv_b + (size_t)l * CONV_DIM, xbc);
        dt_softplus_kernel<<<(MROWS * NHEADS + 255) / 256, blk, 0, stream>>>(
            zx, dt_bias + l * NHEADS, dtb);

        // selective scan
        scan_kernel<<<BATCH * NHEADS, blk, 0, stream>>>(
            xbc, dtb, A_log + l * NHEADS, D_param + l * NHEADS, yb);

        // yz = rmsnorm(y * silu(z), mnw[l]) -> bf16
        gated_rmsnorm_bf16_kernel<<<MROWS, blk, 0, stream>>>(yb, zx, mnw + (size_t)l * D_INNER, yz);

        // x = yz @ out_w[l]^T + x   (residual fused)
        int nWo = D_MODEL * D_INNER;
        cvt_bf16_pad_kernel<<<(nWo + 255) / 256, blk, 0, stream>>>(
            out_w + (size_t)l * nWo, wbf, nWo, nWo);
        gemm_bf16_wmma_kernel<<<dim3(D_MODEL / 64, MROWS / 128), blk, 0, stream>>>(
            yz, wbf, x, x, MROWS, D_MODEL, D_INNER);
    }

    // final norm + lm_head
    rmsnorm_bf16_kernel<<<MROWS, blk, 0, stream>>>(x, norm_f_w, xn, D_MODEL);
    int nWl = VOCAB * D_MODEL;
    cvt_bf16_pad_kernel<<<(nWl + 255) / 256, blk, 0, stream>>>(lm_head, wbf, nWl, nWl);
    gemm_bf16_wmma_kernel<<<dim3(VOCAB / 64, MROWS / 128), blk, 0, stream>>>(
        xn, wbf, (float*)d_out, nullptr, MROWS, VOCAB, D_MODEL);
}
